// LIF_Layer_48258252538647
// MI455X (gfx1250) — matrix operand order repaired and checked by reading, not compiled, not run on hardware
//
#include <hip/hip_runtime.h>

typedef _Float16 half_t;
typedef __attribute__((ext_vector_type(16))) _Float16 v16h;
typedef __attribute__((ext_vector_type(8)))  _Float16 v8h;
typedef __attribute__((ext_vector_type(4)))  _Float16 v4h;
typedef __attribute__((ext_vector_type(8)))  float    v8f;
typedef __attribute__((ext_vector_type(4)))  float    v4f;

#define BATCH 64
#define TIME  256
#define PDIM  1024
#define DDIM  1024
#define MDIM  (BATCH * TIME)   // 16384

// ---------------------------------------------------------------------------
// Phase 1a: spikes f32 -> f16 (values are exactly 0.0/1.0, conversion lossless)
// ---------------------------------------------------------------------------
__global__ __launch_bounds__(256)
void cvt_spikes_f16(const float* __restrict__ in, half_t* __restrict__ out) {
    size_t i = ((size_t)blockIdx.x * blockDim.x + threadIdx.x) * 4;
    v4f v = *(const v4f*)(in + i);
    v4h h;
    h.x = (half_t)v.x; h.y = (half_t)v.y; h.z = (half_t)v.z; h.w = (half_t)v.w;
    *(v4h*)(out + i) = h;
}

// ---------------------------------------------------------------------------
// Phase 1b: weights f32 (P x D, row-major) -> f16 transposed Wt (D x P).
// Transposed storage makes the WMMA B-fragment contiguous in K.
// ---------------------------------------------------------------------------
__global__ __launch_bounds__(256)
void cvt_weights_t_f16(const float* __restrict__ w, half_t* __restrict__ wt) {
    int idx = blockIdx.x * blockDim.x + threadIdx.x;   // 0 .. P*D-1
    int p = idx >> 10;          // row in W
    int d = idx & 1023;         // col in W (coalesced read)
    wt[(size_t)d * PDIM + p] = (half_t)w[idx];
}

// ---------------------------------------------------------------------------
// Phase 2: currents = spikes @ W  via v_wmma_f32_16x16x32_f16.
//   A: MDIM x PDIM f16 row-major (spikes)
//   Bt: DDIM x PDIM f16 row-major (W transposed) -> B[k][n] = Bt[n][k]
//   C: MDIM x DDIM f32
// Block tile 128x128, K-step 32, 8 waves; wave tile 32(M) x 64(N) = 2x4 WMMAs.
//
// Fragment addressing (per CDNA5 ISA, 16-bit layouts, wave32):
//   A 16x32: lane L -> row M = L%16; lanes<16 hold K {kb..kb+7, kb+16..kb+23}
//            with kb = 8*(L/16)  -> two contiguous 8-half (16B) runs.
//   B 32x16: lane L -> col N = L%16; K {kb..kb+7, kb+16..kb+23} as for A, packed 2/VGPR
//            -> two 8-half (16B) runs 16 halves apart = two 16B loads.
// ---------------------------------------------------------------------------
__global__ __launch_bounds__(256)
void lif_gemm_wmma(const half_t* __restrict__ A,
                   const half_t* __restrict__ Bt,
                   float* __restrict__ C) {
    __shared__ __align__(16) half_t sA[128 * 32];   // [m][k]
    __shared__ __align__(16) half_t sB[128 * 32];   // [n][k]

    const int tid  = threadIdx.x;
    const int lane = tid & 31;
    const int wave = tid >> 5;
    const int wm   = wave & 3;          // wave M position (4 waves in M)
    const int wn   = wave >> 2;         // wave N position (2 waves in N)
    const int m0   = blockIdx.y * 128;
    const int n0   = blockIdx.x * 128;

    const int mrow = lane & 15;
    const int khal = lane >> 4;         // 0 or 1: which K-half this lane holds

    v8f acc[2][4];
#pragma unroll
    for (int i = 0; i < 2; ++i)
#pragma unroll
        for (int j = 0; j < 4; ++j) { v8f z = {}; acc[i][j] = z; }

    for (int k0 = 0; k0 < PDIM; k0 += 32) {
        // ---- global -> LDS: 128 rows x 32 halves each, 16B per thread x2 ----
#pragma unroll
        for (int pass = 0; pass < 2; ++pass) {
            int idx = tid + pass * 256;          // 0..511
            int r = idx >> 2;                    // 0..127
            int s = (idx & 3) * 8;               // 0,8,16,24
            *(v8h*)(sA + r * 32 + s) = *(const v8h*)(A  + (size_t)(m0 + r) * PDIM + k0 + s);
            *(v8h*)(sB + r * 32 + s) = *(const v8h*)(Bt + (size_t)(n0 + r) * PDIM + k0 + s);
        }
        __syncthreads();

        // ---- LDS -> fragments ----
        v16h afrag[2], bfrag[4];
#pragma unroll
        for (int i = 0; i < 2; ++i) {
            const half_t* p = sA + (wm * 32 + i * 16 + mrow) * 32 + khal * 8;
            union { v16h v; v8h h[2]; } u;
            u.h[0] = *(const v8h*)(p);        // K = kb .. kb+7
            u.h[1] = *(const v8h*)(p + 16);   // K = kb+16 .. kb+23
            afrag[i] = u.v;
        }
#pragma unroll
        for (int j = 0; j < 4; ++j) {
            const half_t* p = sB + (wn * 64 + j * 16 + mrow) * 32 + khal * 8;
            union { v16h v; v8h h[2]; } u;
            u.h[0] = *(const v8h*)(p);        // K = kb .. kb+7
            u.h[1] = *(const v8h*)(p + 16);   // K = kb+16 .. kb+23
            bfrag[j] = u.v;
        }

        // ---- 8 WMMAs per wave per K-step ----
#pragma unroll
        for (int i = 0; i < 2; ++i)
#pragma unroll
            for (int j = 0; j < 4; ++j)
                acc[i][j] = __builtin_amdgcn_wmma_f32_16x16x32_f16(
                    false, afrag[i], false, bfrag[j],
                    (short)0, acc[i][j], false, false);

        __syncthreads();
    }

    // ---- epilogue: D layout -> lane%16 = N, vgpr r -> M = 8*(lane/16)+r ----
#pragma unroll
    for (int i = 0; i < 2; ++i) {
        int mbase = m0 + wm * 32 + i * 16 + khal * 8;
#pragma unroll
        for (int j = 0; j < 4; ++j) {
            int n = n0 + wn * 64 + j * 16 + mrow;
#pragma unroll
            for (int r = 0; r < 8; ++r)
                C[(size_t)(mbase + r) * DDIM + n] = acc[i][j][r];
        }
    }
}

// ---------------------------------------------------------------------------
// Phase 3: LIF scan over time. One thread per (b, d); coalesced along D.
// voltage output = pre-reset v_new; spike forward value = Heaviside(v-1).
// ---------------------------------------------------------------------------
__global__ __launch_bounds__(256)
void lif_scan(const float* __restrict__ cur,
              float* __restrict__ volt,
              float* __restrict__ spk) {
    int tid = blockIdx.x * blockDim.x + threadIdx.x;   // 0 .. BATCH*DDIM-1
    int b = tid >> 10;
    int d = tid & 1023;
    size_t base = (size_t)b * TIME * DDIM + d;
    float v = 0.0f;
#pragma unroll 4
    for (int t = 0; t < TIME; ++t) {
        size_t idx = base + (size_t)t * DDIM;
        float vn = 0.9f * v + cur[idx];
        float s  = (vn >= 1.0f) ? 1.0f : 0.0f;
        volt[idx] = vn;
        spk[idx]  = s;
        v = vn * (1.0f - s);   // reset-to-zero on spike
    }
}

// ---------------------------------------------------------------------------
extern "C" void kernel_launch(void* const* d_in, const int* in_sizes, int n_in,
                              void* d_out, int out_size, void* d_ws, size_t ws_size,
                              hipStream_t stream) {
    // d_in: [0]=prv_voltage (UNUSED by reference math), [1]=prv_spikes, [2]=weights
    const float* spikes_f32  = (const float*)d_in[1];
    const float* weights_f32 = (const float*)d_in[2];

    // workspace layout:
    //   [0, 32 MiB)           spikes f16  (M x P)
    //   [32 MiB, 34 MiB)      Wt f16      (D x P)
    //   [34 MiB, 98 MiB)      currents f32 (M x D)
    char* ws = (char*)d_ws;
    half_t* sp16 = (half_t*)ws;
    half_t* wt16 = (half_t*)(ws + (size_t)MDIM * PDIM * sizeof(half_t));
    float*  cur  = (float*) (ws + (size_t)MDIM * PDIM * sizeof(half_t)
                                + (size_t)PDIM * DDIM * sizeof(half_t));

    float* volt = (float*)d_out;
    float* spk  = volt + (size_t)BATCH * TIME * DDIM;

    // Phase 1: conversions
    cvt_spikes_f16<<<(MDIM * PDIM) / (256 * 4), 256, 0, stream>>>(spikes_f32, sp16);
    cvt_weights_t_f16<<<(PDIM * DDIM) / 256, 256, 0, stream>>>(weights_f32, wt16);

    // Phase 2: WMMA GEMM  (grid: N-tiles x M-tiles = 8 x 128)
    dim3 ggrid(DDIM / 128, MDIM / 128);
    lif_gemm_wmma<<<ggrid, 256, 0, stream>>>(sp16, wt16, cur);

    // Phase 3: temporal scan
    lif_scan<<<(BATCH * DDIM) / 256, 256, 0, stream>>>(cur, volt, spk);
}
